// SWAdjLoss_21577915695607
// MI455X (gfx1250) — matrix operand order repaired and checked by reading, not compiled, not run on hardware
//
#include <hip/hip_runtime.h>

typedef __attribute__((ext_vector_type(16))) _Float16 v16h;
typedef __attribute__((ext_vector_type(8)))  float    v8f;

namespace {
constexpr int GN  = 256;          // nodes per graph
constexpr int NTH = 1024;         // threads per workgroup (32 wave32)
constexpr unsigned int BIGU = 0x3000u;  // u16 "infinity"; 2*BIGU fits in u16
}

union ABU { unsigned int u[8]; v16h h; };

// One workgroup per graph. Phase A: degrees + WMMA triangle counting on f16
// adjacency in LDS. Phase B: in-place u16 Floyd-Warshall (exact APSP, same
// fixpoint as the reference's 8 min-plus squarings) in the SAME LDS buffer.
__global__ __launch_bounds__(NTH)
void swadj_graph_kernel(const float* __restrict__ pred, float* __restrict__ perGraph) {
  __shared__ __align__(16) _Float16 Af[GN * GN];   // 128 KB, aliased as u16 in phase B
  __shared__ float tri[GN];
  __shared__ float degs[GN];
  __shared__ float clusSum;
  __shared__ unsigned int pathSum, finiteCnt, notConn;

  const int tid = threadIdx.x;
  const int b   = blockIdx.x;
  const float* g = pred + (size_t)b * GN * GN;

  // ---- init shared accumulators ----
  if (tid < GN) { tri[tid] = 0.0f; degs[tid] = 0.0f; }
  if (tid == 0) { clusSum = 0.0f; pathSum = 0u; finiteCnt = 0u; notConn = 0u; }

  // ---- load adjacency (f32 global) -> f16 LDS with zeroed diagonal ----
  {
    const float4* g4 = (const float4*)g;
    #pragma unroll
    for (int c = 0; c < (GN * GN / 4) / NTH; ++c) {
      int e4 = tid + c * NTH;
      float4 fv = g4[e4];
      int e = e4 * 4;
      int i = e >> 8, j = e & (GN - 1);
      fv.x = (j + 0 == i) ? 0.0f : fv.x;     // A = adj - eye
      fv.y = (j + 1 == i) ? 0.0f : fv.y;
      fv.z = (j + 2 == i) ? 0.0f : fv.z;
      fv.w = (j + 3 == i) ? 0.0f : fv.w;
      Af[e + 0] = (_Float16)fv.x;
      Af[e + 1] = (_Float16)fv.y;
      Af[e + 2] = (_Float16)fv.z;
      Af[e + 3] = (_Float16)fv.w;
    }
  }
  __syncthreads();

  // ---- degrees: deg_i = sum_j A[i,j] ----
  {
    int i = tid >> 2, jb = (tid & 3) << 6;
    float s = 0.0f;
    for (int j = 0; j < 64; ++j) s += (float)Af[i * GN + jb + j];
    atomicAdd(&degs[i], s);
  }

  // ---- triangles via WMMA: tri_i += sum_j (A@A)[i,j] * A[i,j] ----
  // 16x16 output-tile grid; wave w owns row-block ti=w/2 and 8 col-blocks,
  // so per-row partial sums stay in registers across col-blocks.
  {
    const int wave = tid >> 5, lane = tid & 31;
    const int ti   = wave >> 1;
    const int half = wave & 1;
    const int i0   = ti * 16;
    const int mlo  = lane & 15;
    const int hi   = (lane >> 4) & 1;

    float rs[8];
    #pragma unroll
    for (int r = 0; r < 8; ++r) rs[r] = 0.0f;

    for (int q = 0; q < 8; ++q) {
      const int j0 = (half * 8 + q) * 16;
      v8f acc = {};
      for (int kb = 0; kb < 8; ++kb) {
        const int k0 = kb * 32;
        ABU a, bm;
        #pragma unroll
        for (int r = 0; r < 8; ++r) {
          // A-tile layout (16-bit A 16x32): lanes0-15 K=0..7/16..23, lanes16-31 +8
          int koffA = ((r < 4) ? 2 * r : 16 + 2 * (r - 4)) + (hi ? 8 : 0);
          a.u[r]  = *(const unsigned int*)&Af[(i0 + mlo) * GN + k0 + koffA];
          // B-tile: B[k][n] = A[k][j0+n] = A[j0+n][k] (symmetric) -> contiguous pair
          int koffB = ((r < 4) ? 2 * r : 16 + 2 * (r - 4)) + (hi ? 8 : 0);
          bm.u[r] = *(const unsigned int*)&Af[(j0 + mlo) * GN + k0 + koffB];
        }
        acc = __builtin_amdgcn_wmma_f32_16x16x32_f16(
                  false, a.h, false, bm.h, (short)0, acc, false, false);
      }
      #pragma unroll
      for (int r = 0; r < 8; ++r) {
        int m = r + (hi ? 8 : 0);        // C/D layout: VGPR r, lanes>=16 -> M+8
        rs[r] += acc[r] * (float)Af[(i0 + m) * GN + j0 + mlo];
      }
    }
    #pragma unroll
    for (int r = 0; r < 8; ++r) {
      int m = r + (hi ? 8 : 0);
      atomicAdd(&tri[i0 + m], rs[r]);
    }
  }
  __syncthreads();

  // ---- clustering coefficient reduction ----
  if (tid < GN) {
    float dg   = degs[tid];
    float t    = 0.5f * tri[tid];
    float poss = dg * (dg - 1.0f) * 0.5f;
    float cc   = (dg > 1.0f) ? (t / ((poss > 0.0f) ? poss : 1.0f)) : 0.0f;
    atomicAdd(&clusSum, cc);
  }

  // ---- in-place convert f16 adjacency -> u16 distance init (same LDS bytes) ----
  unsigned short* Du = (unsigned short*)Af;
  {
    int i = tid >> 2, jb = (tid & 3) << 6;
    for (int j = 0; j < 64; ++j) {
      int e = i * GN + jb + j;
      float av = (float)Af[e];
      Du[e] = (av > 0.5f) ? (unsigned short)1 : (unsigned short)BIGU;
    }
  }
  __syncthreads();

  // ---- in-place Floyd-Warshall (exact APSP fixpoint; packed u16 min-plus) ----
  // Row k / column k are numerically invariant during step k, so concurrent
  // reads are race-free; one barrier per k publishes the step.
  {
    const int i  = tid >> 2;
    const int jb = (tid & 3) << 6;
    uint4* dr = (uint4*)(Du + i * GN + jb);
    for (int k = 0; k < GN; ++k) {
      unsigned int dik = (unsigned int)Du[i * GN + k];
      unsigned int ddk = dik * 0x00010001u;   // packed {dik,dik}
      const uint4* rk = (const uint4*)(Du + k * GN + jb);
      #pragma unroll
      for (int c = 0; c < 8; ++c) {
        uint4 rv = rk[c];
        uint4 dv = dr[c];
        unsigned int s0, s1, s2, s3, m0, m1, m2, m3;
        asm("v_pk_add_u16 %0, %1, %2" : "=v"(s0) : "v"(rv.x), "v"(ddk));
        asm("v_pk_add_u16 %0, %1, %2" : "=v"(s1) : "v"(rv.y), "v"(ddk));
        asm("v_pk_add_u16 %0, %1, %2" : "=v"(s2) : "v"(rv.z), "v"(ddk));
        asm("v_pk_add_u16 %0, %1, %2" : "=v"(s3) : "v"(rv.w), "v"(ddk));
        asm("v_pk_min_u16 %0, %1, %2" : "=v"(m0) : "v"(s0), "v"(dv.x));
        asm("v_pk_min_u16 %0, %1, %2" : "=v"(m1) : "v"(s1), "v"(dv.y));
        asm("v_pk_min_u16 %0, %1, %2" : "=v"(m2) : "v"(s2), "v"(dv.z));
        asm("v_pk_min_u16 %0, %1, %2" : "=v"(m3) : "v"(s3), "v"(dv.w));
        dv.x = m0; dv.y = m1; dv.z = m2; dv.w = m3;
        dr[c] = dv;
      }
      __syncthreads();
    }
  }

  // ---- path-length reduction + connectivity (row 0 all finite) ----
  {
    const int i = tid >> 2, jb = (tid & 3) << 6;
    unsigned int ls = 0, lc = 0, bad = 0;
    for (int j = 0; j < 64; ++j) {
      unsigned int d = (unsigned int)Du[i * GN + jb + j];
      if (d < BIGU) { ls += d; lc++; }
      else if (i == 0) bad = 1u;
    }
    atomicAdd(&pathSum, ls);
    atomicAdd(&finiteCnt, lc);
    if (bad) atomicOr(&notConn, 1u);
  }
  __syncthreads();

  if (tid == 0) {
    bool conn = (notConn == 0u);
    float avgClus = clusSum * (1.0f / (float)GN);
    float avgPath = (finiteCnt > 0u) ? ((float)pathSum / (float)finiteCnt) : 0.0f;
    // NOTE: the reference swaps these names on purpose; reproduced faithfully.
    float pred_apl = conn ? avgClus : (float)GN;   // vs target log2(256)=8
    float pred_cc  = conn ? avgPath : 0.0f;        // vs target 0.8
    float d1 = pred_apl - 8.0f;
    float d2 = pred_cc  - 0.8f;
    perGraph[b] = d1 * d1 + d2 * d2;
  }
}

__global__ void swadj_reduce_kernel(const float* __restrict__ perGraph,
                                    float* __restrict__ out, int B) {
  if (threadIdx.x == 0) {
    float s = 0.0f;
    for (int i = 0; i < B; ++i) s += perGraph[i];
    out[0] = s / (float)B;
  }
}

extern "C" void kernel_launch(void* const* d_in, const int* in_sizes, int n_in,
                              void* d_out, int out_size, void* d_ws, size_t ws_size,
                              hipStream_t stream) {
  const float* pred = (const float*)d_in[0];
  float* out = (float*)d_out;
  float* ws  = (float*)d_ws;                 // B per-graph loss contributions
  int B = in_sizes[0] / (GN * GN);
  swadj_graph_kernel<<<B, NTH, 0, stream>>>(pred, ws);
  swadj_reduce_kernel<<<1, 32, 0, stream>>>(ws, out, B);
}
